// MaxSigmoidAttnBlock_1494648619580
// MI455X (gfx1250) — compile-verified, matrix operand order checked by reading, not run on hardware
//
#include <hip/hip_runtime.h>

typedef __attribute__((ext_vector_type(16))) _Float16 v16h;
typedef __attribute__((ext_vector_type(8)))  float    v8f;

#define BS  8
#define C1  256
#define EC  512
#define NH  8
#define HC  64
#define H   64
#define W   64
#define HW  4096
#define NTOK 80
#define GC  512

// Fragment = two contiguous 16B runs (ISA 7.12.2 16-bit 16x32 A layout, group g):
//   halves [g*8 .. g*8+7] and [16+g*8 .. 16+g*8+7] of a 32-half row.
union FragH { int4 q[2]; v16h v; };

__device__ __forceinline__ v16h load_frag(const _Float16* rowbase, int g) {
    FragH f;
    f.q[0] = *reinterpret_cast<const int4*>(rowbase + g * 8);
    f.q[1] = *reinterpret_cast<const int4*>(rowbase + 16 + g * 8);
    return f.v;
}
__device__ __forceinline__ v16h zero_frag() {
    FragH f;
    f.q[0] = make_int4(0, 0, 0, 0);
    f.q[1] = make_int4(0, 0, 0, 0);
    return f.v;
}
__device__ __forceinline__ void copy8h(_Float16* dst, const _Float16* src) {
    *reinterpret_cast<int4*>(dst) = *reinterpret_cast<const int4*>(src);
}

// ------------------------------------------------------------------
// Kernel 1: fold BN into conv weights, convert to f16.
__global__ void prep_weights(const float* __restrict__ ec_w, const float* __restrict__ ec_g,
                             const float* __restrict__ ec_b, const float* __restrict__ ec_m,
                             const float* __restrict__ ec_v,
                             const float* __restrict__ pj_w, const float* __restrict__ pj_g,
                             const float* __restrict__ pj_b, const float* __restrict__ pj_m,
                             const float* __restrict__ pj_v,
                             _Float16* __restrict__ w1h, float* __restrict__ bias1,
                             _Float16* __restrict__ w3h, float* __restrict__ bias3) {
    int e = blockIdx.x, c = threadIdx.x;
    float s1 = ec_g[e] * rsqrtf(ec_v[e] + 1e-3f);
    float s3 = pj_g[e] * rsqrtf(pj_v[e] + 1e-3f);
    w1h[e * C1 + c] = (_Float16)(ec_w[e * C1 + c] * s1);
#pragma unroll
    for (int tap = 0; tap < 9; ++tap)
        w3h[((size_t)tap * EC + e) * C1 + c] = (_Float16)(pj_w[((size_t)e * C1 + c) * 9 + tap] * s3);
    if (c == 0) {
        bias1[e] = ec_b[e] - ec_m[e] * s1;
        bias3[e] = pj_b[e] - pj_m[e] * s3;
    }
}

// ------------------------------------------------------------------
// Kernel 2: x f32 -> f16 (NCHW preserved).
__global__ void convert_x(const float* __restrict__ x, _Float16* __restrict__ xh) {
    size_t i = ((size_t)blockIdx.x * blockDim.x + threadIdx.x) * 4;
    float4 v = *reinterpret_cast<const float4*>(x + i);
    union { _Float16 h[4]; uint2 u; } p;
    p.h[0] = (_Float16)v.x; p.h[1] = (_Float16)v.y;
    p.h[2] = (_Float16)v.z; p.h[3] = (_Float16)v.w;
    *reinterpret_cast<uint2*>(xh + i) = p.u;
}

// ------------------------------------------------------------------
// Kernel 3: guide linear  gf[b][n][e] = guide[b,n,:] . gl_w[e,:] + gl_b[e]
__global__ void guide_linear(const float* __restrict__ guide, const float* __restrict__ gl_w,
                             const float* __restrict__ gl_b, float* __restrict__ gf) {
    int e = blockIdx.x * 256 + threadIdx.x;
    int n = blockIdx.y, b = blockIdx.z;
    const float4* gp = reinterpret_cast<const float4*>(guide + ((size_t)b * NTOK + n) * GC);
    const float4* wp = reinterpret_cast<const float4*>(gl_w + (size_t)e * GC);
    float s = 0.f;
#pragma unroll 4
    for (int i = 0; i < GC / 4; ++i) {
        float4 a = gp[i], w = wp[i];
        s += a.x * w.x + a.y * w.y + a.z * w.z + a.w * w.w;
    }
    gf[((size_t)b * NTOK + n) * EC + e] = s + gl_b[e];
}

// ------------------------------------------------------------------
// Stage 32ch x 64px transposed into LDS as [px][kk] (row stride 32 halves).
__device__ __forceinline__ void stage_xT64(_Float16* __restrict__ lxT,
                                           const _Float16* __restrict__ src_row0,
                                           int t) {
    const int kk0 = (t & 15) * 2;
    const int px0 = (t >> 4) * 4;
    union U4 { _Float16 h[4]; uint2 u; };
    U4 a, b;
    a.u = *reinterpret_cast<const uint2*>(src_row0 + (size_t)kk0 * HW + px0);
    b.u = *reinterpret_cast<const uint2*>(src_row0 + (size_t)(kk0 + 1) * HW + px0);
#pragma unroll
    for (int i = 0; i < 4; ++i) {
        union { _Float16 h[2]; unsigned u; } p;
        p.h[0] = a.h[i];
        p.h[1] = b.h[i];
        *reinterpret_cast<unsigned*>(&lxT[(px0 + i) * 32 + kk0]) = p.u;
    }
}
__device__ __forceinline__ void stage_xT64_zero(_Float16* __restrict__ lxT, int t) {
    const int kk0 = (t & 15) * 2;
    const int px0 = (t >> 4) * 4;
#pragma unroll
    for (int i = 0; i < 4; ++i)
        *reinterpret_cast<unsigned*>(&lxT[(px0 + i) * 32 + kk0]) = 0u;
}

// Stage 32ch x 128px transposed (1x1 kernel's wider tile).
__device__ __forceinline__ void stage_xT128(_Float16* __restrict__ lxT,
                                            const _Float16* __restrict__ src_row0,
                                            int t) {
    const int kk0 = (t & 15) * 2;
    const int px0 = (t >> 4) * 8;
    union U8 { _Float16 h[8]; int4 q; };
    U8 a, b;
    a.q = *reinterpret_cast<const int4*>(src_row0 + (size_t)kk0 * HW + px0);
    b.q = *reinterpret_cast<const int4*>(src_row0 + (size_t)(kk0 + 1) * HW + px0);
#pragma unroll
    for (int i = 0; i < 8; ++i) {
        union { _Float16 h[2]; unsigned u; } p;
        p.h[0] = a.h[i];
        p.h[1] = b.h[i];
        *reinterpret_cast<unsigned*>(&lxT[(px0 + i) * 32 + kk0]) = p.u;
    }
}

// ------------------------------------------------------------------
// Kernel 4: 1x1 conv (WMMA) + BN + max-over-tokens + sigmoid gate.
// Block = 2 heads (128 ch) x 128 px. 8 waves; wave = (32-ch block mq, 64-px half).
// Per k-step: 2 A frags share 4 B frags -> 8 WMMA (12 ds_load_b128 : 8 wmma).
// grid (HW/128, NH/2, BS), 256 thr.
__global__ void conv1x1_attn(const _Float16* __restrict__ xh, const _Float16* __restrict__ w1h,
                             const float* __restrict__ bias1, const float* __restrict__ gf,
                             const float* __restrict__ attn_bias, const float* __restrict__ attn_scale,
                             float* __restrict__ awbuf) {
    const int pb = blockIdx.x * 128;
    const int b = blockIdx.z;
    const int e0 = blockIdx.y * 128;       // first of two heads' channel base

    __shared__ _Float16 lw[128 * 32];      // A stage [m][kk]          8 KB
    __shared__ _Float16 lxT[128 * 32];     // B stage transposed       8 KB
    __shared__ _Float16 lembh[128 * 128];  // emb tile f16 [c][px]    32 KB

    const int t = threadIdx.x;
    const int lane = t & 31;
    const int wv = t >> 5;
    const int mq = wv >> 1;                // 0..3 : 32-channel block
    const int colbase = (wv & 1) * 64;

    const int grp = lane >> 4;
    const int cn = lane & 15;
    const int mr0 = mq * 32 + cn;
    const int mr1 = mr0 + 16;

    v8f acc[2][4] = {{{}, {}, {}, {}}, {{}, {}, {}, {}}};

    for (int k0 = 0; k0 < C1; k0 += 32) {
        __syncthreads();
        { // stage A: 128x32 halves row-major (16 halves per thread)
            int m = t >> 1, kkb = (t & 1) * 16;
            const _Float16* src = w1h + (size_t)(e0 + m) * C1 + k0 + kkb;
            copy8h(&lw[m * 32 + kkb], src);
            copy8h(&lw[m * 32 + kkb + 8], src + 8);
        }
        stage_xT128(lxT, xh + ((size_t)b * C1 + k0) * HW + pb, t);
        __syncthreads();

        v16h a0 = load_frag(&lw[mr0 * 32], grp);
        v16h a1 = load_frag(&lw[mr1 * 32], grp);
#pragma unroll
        for (int nb = 0; nb < 4; ++nb) {
            v16h bf = load_frag(&lxT[(colbase + nb * 16 + cn) * 32], grp);
            acc[0][nb] = __builtin_amdgcn_wmma_f32_16x16x32_f16(false, a0, false, bf, (short)0, acc[0][nb], false, false);
            acc[1][nb] = __builtin_amdgcn_wmma_f32_16x16x32_f16(false, a1, false, bf, (short)0, acc[1][nb], false, false);
        }
    }

    // C layout: lanes0-15 N=lane, M=r ; lanes16-31 N=lane-16, M=8+r
    {
        const int mhi = (lane >> 4) * 8;
#pragma unroll
        for (int i = 0; i < 2; ++i)
#pragma unroll
            for (int r = 0; r < 8; ++r) {
                int m = mq * 32 + i * 16 + mhi + r;
                float bs1 = bias1[e0 + m];
#pragma unroll
                for (int nb = 0; nb < 4; ++nb)
                    lembh[m * 128 + colbase + nb * 16 + cn] = (_Float16)(acc[i][nb][r] + bs1);
            }
    }
    __syncthreads();

    // attention: thread -> (head-half, pixel); full 80-token max per thread (no reduction needed)
    const int px = t & 127, hh = t >> 7;
    const int head = blockIdx.y * 2 + hh;
    const _Float16* eh = lembh + hh * 64 * 128;
    const float* gofs = gf + ((size_t)b * NTOK) * EC + (size_t)head * HC;
    float mx = -3.4e38f;
    for (int n = 0; n < NTOK; ++n) {
        const float* gn = gofs + (size_t)n * EC;
        float s = 0.f;
#pragma unroll 8
        for (int c = 0; c < HC; ++c) s += (float)eh[c * 128 + px] * gn[c];
        mx = fmaxf(mx, s);
    }
    float logit = mx * 0.125f + attn_bias[head];           // 1/sqrt(64)
    float sig = 1.f / (1.f + __expf(-logit));
    awbuf[((size_t)b * NH + head) * HW + pb + px] = sig * attn_scale[head];
}

// ------------------------------------------------------------------
// Kernel 5: 3x3 conv (implicit-GEMM WMMA) + BN + per-head gate.
// Block = 128 ch x 2 output rows. 8 waves; wave = (32-ch block mq, output-row select).
// Weights staged per-kh group (3 taps, 24 KB). Per tap: 2 A frags share 4 B frags -> 8 WMMA.
// grid (H/2, EC/128, BS), 256 thr.
__global__ void conv3x3_gate(const _Float16* __restrict__ xh, const _Float16* __restrict__ w3h,
                             const float* __restrict__ bias3, const float* __restrict__ awbuf,
                             float* __restrict__ out) {
    const int hp = blockIdx.x, b = blockIdx.z;
    const int e0 = blockIdx.y * 128;
    const int rbase = hp * 2 - 1;           // first staged input row

    __shared__ _Float16 lxT[4 * 64 * 32];   // [row rr][px][kk]        16 KB
    __shared__ _Float16 lwv[3 * 128 * 32];  // [kw][m][kk] current kh  24 KB

    const int t = threadIdx.x;
    const int lane = t & 31;
    const int wv = t >> 5;
    const int mq = wv >> 1;                 // 0..3 : 32-channel block
    const int hout = hp * 2 + (wv & 1);     // this wave's output row

    const int grp = lane >> 4;
    const int cn = lane & 15;
    const int mr0 = mq * 32 + cn;
    const int mr1 = mr0 + 16;

    v8f acc[2][4] = {{{}, {}, {}, {}}, {{}, {}, {}, {}}};
    const v16h zf = zero_frag();

    for (int k0 = 0; k0 < C1; k0 += 32) {
#pragma unroll
        for (int kh = 0; kh < 3; ++kh) {
            __syncthreads();
            if (kh == 0) { // stage 4 input rows (transposed), zero out-of-image rows
#pragma unroll
                for (int rr = 0; rr < 4; ++rr) {
                    int row = rbase + rr;
                    if (row >= 0 && row < H) {
                        const _Float16* src = xh + (((size_t)b * C1 + k0) * H + row) * W;
                        stage_xT64(&lxT[rr * 64 * 32], src, t);
                        if (k0 + 32 < C1)
                            __builtin_prefetch(src + (size_t)32 * HW, 0, 0);
                    } else {
                        stage_xT64_zero(&lxT[rr * 64 * 32], t);
                    }
                }
            }
            { // stage this kh-group's 3 taps of A (128x32 each)
                int m = t >> 1, kkb = (t & 1) * 16;
#pragma unroll
                for (int kw = 0; kw < 3; ++kw) {
                    const _Float16* src = w3h + ((size_t)(kh * 3 + kw) * EC + e0 + m) * C1 + k0 + kkb;
                    copy8h(&lwv[(kw * 128 + m) * 32 + kkb], src);
                    copy8h(&lwv[(kw * 128 + m) * 32 + kkb + 8], src + 8);
                }
            }
            __syncthreads();

            const int row = hout + kh - 1;
            if (row < 0 || row >= H) continue;           // wave-uniform, no barriers below
            const int rr = row - rbase;                  // 0..3
#pragma unroll
            for (int kw = 0; kw < 3; ++kw) {
                v16h a0 = load_frag(&lwv[(kw * 128 + mr0) * 32], grp);
                v16h a1 = load_frag(&lwv[(kw * 128 + mr1) * 32], grp);
#pragma unroll
                for (int nb = 0; nb < 4; ++nb) {
                    const int w0 = nb * 16 + cn + kw - 1;    // [-1, 64]
                    v16h bf = (w0 >= 0 && w0 < W) ? load_frag(&lxT[(rr * 64 + w0) * 32], grp) : zf;
                    acc[0][nb] = __builtin_amdgcn_wmma_f32_16x16x32_f16(false, a0, false, bf, (short)0, acc[0][nb], false, false);
                    acc[1][nb] = __builtin_amdgcn_wmma_f32_16x16x32_f16(false, a1, false, bf, (short)0, acc[1][nb], false, false);
                }
            }
        }
    }

    // epilogue: BN bias + per-head gate, write NCHW
    const int head = blockIdx.y * 2 + (mq >> 1);         // channel block -> head
    const float* aw = awbuf + ((size_t)b * NH + head) * HW + (size_t)hout * W;
    const int mhi = (lane >> 4) * 8;
#pragma unroll
    for (int i = 0; i < 2; ++i)
#pragma unroll
        for (int r = 0; r < 8; ++r) {
            int m = mq * 32 + i * 16 + mhi + r;
            float bs3 = bias3[e0 + m];
            size_t o = (((size_t)b * EC + e0 + m) * H + hout) * W;
#pragma unroll
            for (int nb = 0; nb < 4; ++nb) {
                int px = nb * 16 + cn;
                out[o + px] = (acc[i][nb][r] + bs3) * aw[px];
            }
        }
}

// ------------------------------------------------------------------
extern "C" void kernel_launch(void* const* d_in, const int* in_sizes, int n_in,
                              void* d_out, int out_size, void* d_ws, size_t ws_size,
                              hipStream_t stream) {
    const float* x          = (const float*)d_in[0];
    const float* guide      = (const float*)d_in[1];
    const float* ec_w       = (const float*)d_in[2];
    const float* ec_g       = (const float*)d_in[3];
    const float* ec_b       = (const float*)d_in[4];
    const float* ec_m       = (const float*)d_in[5];
    const float* ec_v       = (const float*)d_in[6];
    const float* gl_w       = (const float*)d_in[7];
    const float* gl_b       = (const float*)d_in[8];
    const float* attn_bias  = (const float*)d_in[9];
    const float* attn_scale = (const float*)d_in[10];
    const float* pj_w       = (const float*)d_in[11];
    const float* pj_g       = (const float*)d_in[12];
    const float* pj_b       = (const float*)d_in[13];
    const float* pj_m       = (const float*)d_in[14];
    const float* pj_v       = (const float*)d_in[15];
    float* out = (float*)d_out;

    char* ws = (char*)d_ws;
    size_t off = 0;
    auto take = [&](size_t bytes) { size_t o = off; off = (off + bytes + 255) & ~(size_t)255; return o; };
    _Float16* w1h  = ( _Float16*)(ws + take((size_t)EC * C1 * 2));
    float*    bias1= (    float*)(ws + take((size_t)EC * 4));
    _Float16* w3h  = ( _Float16*)(ws + take((size_t)9 * EC * C1 * 2));
    float*    bias3= (    float*)(ws + take((size_t)EC * 4));
    _Float16* xh   = ( _Float16*)(ws + take((size_t)BS * C1 * HW * 2));
    float*    gf   = (    float*)(ws + take((size_t)BS * NTOK * EC * 4));
    float*    awb  = (    float*)(ws + take((size_t)BS * NH * HW * 4));

    prep_weights<<<EC, C1, 0, stream>>>(ec_w, ec_g, ec_b, ec_m, ec_v,
                                        pj_w, pj_g, pj_b, pj_m, pj_v,
                                        w1h, bias1, w3h, bias3);
    convert_x<<<(BS * C1 * HW) / (256 * 4), 256, 0, stream>>>(x, xh);
    guide_linear<<<dim3(EC / 256, NTOK, BS), 256, 0, stream>>>(guide, gl_w, gl_b, gf);
    conv1x1_attn<<<dim3(HW / 128, NH / 2, BS), 256, 0, stream>>>(xh, w1h, bias1, gf,
                                                                 attn_bias, attn_scale, awb);
    conv3x3_gate<<<dim3(H / 2, EC / 128, BS), 256, 0, stream>>>(xh, w3h, bias3, awb, out);
}
